// Block_2808908611837
// MI455X (gfx1250) — matrix operand order repaired and checked by reading, not compiled, not run on hardware
//
#include <hip/hip_runtime.h>
#include <hip/hip_bf16.h>
#include <cstdint>

typedef __attribute__((ext_vector_type(16))) _Float16 v16h;
typedef __attribute__((ext_vector_type(8)))  _Float16 v8h;
typedef __attribute__((ext_vector_type(8)))  float    v8f;
typedef __attribute__((ext_vector_type(4)))  unsigned int u32x4;
typedef __attribute__((ext_vector_type(8)))  int      i32x8;
typedef __attribute__((ext_vector_type(4)))  int      i32x4;

#define DIMC   768
#define HEADS  12
#define HDIM   64
#define HIDDEN 3072
#define SEQ    2048
#define BATCH  2
#define TOKENS (BATCH*SEQ)
#define LNEPS  1e-5f
#define ATSCALE 0.125f   // 64^-0.5

#define HAVE_TDM 0
#if defined(__has_builtin)
#if __has_builtin(__builtin_amdgcn_tensor_load_to_lds) && __has_builtin(__builtin_amdgcn_s_wait_tensorcnt)
#undef HAVE_TDM
#define HAVE_TDM 1
#endif
#endif

// ---------------------------------------------------------------- helpers
__device__ inline v8f wmma_f16(v16h a, v16h b, v8f c) {
    return __builtin_amdgcn_wmma_f32_16x16x32_f16(
        /*neg_a=*/false, a, /*neg_b=*/false, b,
        /*c_mod=*/(short)0, c, /*reuse_a=*/false, /*reuse_b=*/false);
}

// A-fragment (16x32 f16, MxK): lane holds row M=lane%16; elements 0..15 are
// K = g*16..g*16+15 (g = lane/16).
__device__ inline v16h load_a_frag(const _Float16* __restrict__ row, int k0, int g) {
    v8h lo = *(const v8h*)(row + k0 + g * 16);
    v8h hi = *(const v8h*)(row + k0 + g * 16 + 8);
    v16h a;
#pragma unroll
    for (int i = 0; i < 8; ++i) { a[i] = lo[i]; a[8 + i] = hi[i]; }
    return a;
}

#if HAVE_TDM
// Write `v` into bits [lo, lo+w) of a dword array (TDM descriptor packing).
__device__ inline void set_bits(uint32_t* d, int lo, int w, unsigned long long v) {
    int idx = lo >> 5, sh = lo & 31;
    unsigned long long cur = v << sh;
    d[idx] |= (uint32_t)cur;
    if (sh + w > 32) d[idx + 1] |= (uint32_t)(cur >> 32);
}

__device__ inline uint32_t lds_offset(const void* p) {
    return (uint32_t)(uintptr_t)p;   // flat shared addr: LDS offset = addr[31:0]
}

// Issue TENSOR_LOAD_TO_LDS for a 2D tile: 64 rows x 64 f16 (row stride K elems).
// D# layout per CDNA5 ISA.
__device__ inline void tdm_load_b_tile(const _Float16* src, int strideElems,
                                       int remRows, int remK, uint32_t ldsOff) {
    uint32_t d0[4] = {0, 0, 0, 0};
    uint32_t d1[8] = {0, 0, 0, 0, 0, 0, 0, 0};
    set_bits(d0, 0, 2, 1);                                        // count = 1 (valid)
    set_bits(d0, 32, 32, ldsOff);                                 // lds_addr
    set_bits(d0, 64, 57, (unsigned long long)(uintptr_t)src);     // global_addr (tile start)
    set_bits(d0, 126, 2, 2);                                      // type = 2 ("image")
    set_bits(d1, 16, 2, 1);                                       // data_size = 2 bytes
    set_bits(d1, 48, 32, (unsigned)remK);                         // tensor_dim0
    set_bits(d1, 80, 32, (unsigned)remRows);                      // tensor_dim1
    set_bits(d1, 112, 16, 64);                                    // tile_dim0 (k elems, 128B row)
    set_bits(d1, 128, 16, 64);                                    // tile_dim1 (n rows)
    set_bits(d1, 160, 48, (unsigned)strideElems);                 // tensor_dim0_stride
    u32x4 g0; i32x8 g1;
#pragma unroll
    for (int i = 0; i < 4; ++i) g0[i] = d0[i];
#pragma unroll
    for (int i = 0; i < 8; ++i) g1[i] = (int)d1[i];
    i32x4 z4 = {0, 0, 0, 0};
#if __clang_major__ >= 23
    i32x8 z8 = {0, 0, 0, 0, 0, 0, 0, 0};
    __builtin_amdgcn_tensor_load_to_lds(g0, g1, z4, z4, z8, 0);
#else
    __builtin_amdgcn_tensor_load_to_lds(g0, g1, z4, z4, 0);
#endif
}
#endif  // HAVE_TDM

// ---------------------------------------------------------------- cast
__global__ void cast_f32_to_f16(const float* __restrict__ in,
                                _Float16* __restrict__ out, int n) {
    int i = blockIdx.x * blockDim.x + threadIdx.x;
    int stride = gridDim.x * blockDim.x;
    for (; i < n; i += stride) out[i] = (_Float16)in[i];
}

// ---------------------------------------------------------------- layernorm
__global__ void layernorm_to_f16(const float* __restrict__ x,
                                 const float* __restrict__ gam,
                                 const float* __restrict__ bet,
                                 _Float16* __restrict__ out) {
    __shared__ float red[256];
    int row = blockIdx.x;
    const float* xr = x + (size_t)row * DIMC;
    float s = 0.f;
    for (int i = threadIdx.x; i < DIMC; i += 256) s += xr[i];
    red[threadIdx.x] = s; __syncthreads();
    for (int off = 128; off > 0; off >>= 1) {
        if (threadIdx.x < off) red[threadIdx.x] += red[threadIdx.x + off];
        __syncthreads();
    }
    float mean = red[0] * (1.0f / DIMC);
    __syncthreads();
    float v = 0.f;
    for (int i = threadIdx.x; i < DIMC; i += 256) { float d = xr[i] - mean; v += d * d; }
    red[threadIdx.x] = v; __syncthreads();
    for (int off = 128; off > 0; off >>= 1) {
        if (threadIdx.x < off) red[threadIdx.x] += red[threadIdx.x + off];
        __syncthreads();
    }
    float rstd = rsqrtf(red[0] * (1.0f / DIMC) + LNEPS);
    for (int i = threadIdx.x; i < DIMC; i += 256)
        out[(size_t)row * DIMC + i] = (_Float16)((xr[i] - mean) * rstd * gam[i] + bet[i]);
}

// ---------------------------------------------------------------- V transpose
// qkv[token][2304] (v at +1536) -> vT[((b*H+h)*64+d)*SEQ + token]  (f16)
__global__ void transpose_v(const _Float16* __restrict__ qkv,
                            _Float16* __restrict__ vT) {
    int idx = blockIdx.x * 256 + threadIdx.x;   // over B*H*64*SEQ elements
    int tok = idx & (SEQ - 1);
    int rest = idx >> 11;
    int d = rest & 63;
    int bh = rest >> 6;
    int h = bh % HEADS, b = bh / HEADS;
    vT[idx] = qkv[((size_t)(b * SEQ + tok)) * 2304 + 1536 + h * 64 + d];
}

// ---------------------------------------------------------------- GEMM
// out[M,N] = A[M,K] @ W[N,K]^T (+bias) (gelu?) (+res); out f16 or f32.
// Block = 4 waves = 64x64 output tile. The 64xK weight slab is shared by all
// 4 waves: staged 64n x 64k f16 (8KB) into LDS via TDM (tensor_load_to_lds),
// double-buffered, TENSORcnt-tracked; waves consume it via ds_load_b128.
__global__ void gemm_wmma(const _Float16* __restrict__ A,
                          const _Float16* __restrict__ W,
                          const float* __restrict__ bias,
                          const float* __restrict__ res,
                          float* __restrict__ out32,
                          _Float16* __restrict__ out16,
                          int M, int N, int K, int gelu) {
    __shared__ _Float16 lds_b[2][64 * 64];
    int wave = threadIdx.x >> 5;
    int lane = threadIdx.x & 31;
    int g = lane >> 4, c = lane & 15;
    int ntiles = N >> 6;
    int bx = blockIdx.x % ntiles;
    int by = blockIdx.x / ntiles;
    int m0 = by * 64 + wave * 16;
    int n0 = bx * 64;

    const v8f z8 = {0.f,0.f,0.f,0.f,0.f,0.f,0.f,0.f};
    v8f acc[4] = {z8, z8, z8, z8};

    const _Float16* arow = A + (size_t)(m0 + c) * K;

#if HAVE_TDM
    if (wave == 0)
        tdm_load_b_tile(W + (size_t)n0 * K, K, N - n0, K,
                        lds_offset(&lds_b[0][0]));
#endif

    for (int kb = 0; kb < K; kb += 64) {
#if HAVE_TDM
        int buf = (kb >> 6) & 1;
        if (wave == 0) __builtin_amdgcn_s_wait_tensorcnt(0);
        __syncthreads();                       // tile `buf` ready for all waves
        if (wave == 0 && kb + 64 < K)          // prefetch next slab into buf^1
            tdm_load_b_tile(W + (size_t)n0 * K + kb + 64, K,
                            N - n0, K - (kb + 64),
                            lds_offset(&lds_b[buf ^ 1][0]));
        const _Float16* bb = lds_b[buf];
#else
        __syncthreads();
        {   // cooperative staging fallback: 128 threads x 64B
            int t = threadIdx.x;
            int n = t >> 1, hk = t & 1;
            const v16h* sp = (const v16h*)(W + (size_t)(n0 + n) * K + kb + hk * 32);
            v16h* dp = (v16h*)&lds_b[0][n * 64 + hk * 32];
            dp[0] = sp[0]; dp[1] = sp[1];
        }
        __syncthreads();
        const _Float16* bb = lds_b[0];
#endif
        __builtin_prefetch(arow + kb + 128, 0, 1);   // global_prefetch_b8 (A stream)
#pragma unroll
        for (int ks = 0; ks < 64; ks += 32) {
            v16h a = load_a_frag(arow, kb + ks, g);
            v16h bfr[4];
#pragma unroll
            for (int nt = 0; nt < 4; ++nt)   // hoist all ds loads: one wait,
                bfr[nt] = *(const v16h*)(bb + (size_t)(nt * 16 + c) * 64 + ks + g * 16);
#pragma unroll
            for (int nt = 0; nt < 4; ++nt)   // then back-to-back WMMAs
                acc[nt] = wmma_f16(a, bfr[nt], acc[nt]);
        }
    }

#pragma unroll
    for (int nt = 0; nt < 4; ++nt) {
        int n = n0 + nt * 16 + c;
        float bv = bias ? bias[n] : 0.f;
#pragma unroll
        for (int r = 0; r < 8; ++r) {
            int m = m0 + r + 8 * g;
            float val = acc[nt][r] + bv;
            if (gelu) val = 0.5f * val * (1.0f + erff(val * 0.70710678118f));
            if (res)  val += res[(size_t)m * N + n];
            if (out16) out16[(size_t)m * N + n] = (_Float16)val;
            else       out32[(size_t)m * N + n] = val;
        }
    }
}

// ---------------------------------------------------------------- attention
// One wave per 16-query tile per (b,h). Online softmax over 32-key steps.
// qkv layout per token row (2304 f16): [q(12x64) | k(12x64) | v(12x64)].
// vT layout: [((b*H+h)*64+d)*SEQ + token] so PV B-fragments are contiguous.
__global__ void attn_wmma(const _Float16* __restrict__ qkv,
                          const _Float16* __restrict__ vT,
                          const int* __restrict__ mask,
                          _Float16* __restrict__ y) {
    __shared__ _Float16 pl[16 * 32];
    int lane = threadIdx.x & 31;
    int g = lane >> 4, c = lane & 15;
    int qt = blockIdx.x & 127;          // SEQ/16 = 128
    int bh = blockIdx.x >> 7;
    int h = bh % HEADS, b = bh / HEADS;

    const _Float16* qbase = qkv + (size_t)b * SEQ * 2304 + h * 64;
    const _Float16* kbase = qbase + 768;
    const _Float16* vtb   = vT + (size_t)bh * HDIM * SEQ;   // row d: vtb + d*SEQ

    const _Float16* qrow = qbase + (size_t)(qt * 16 + c) * 2304;
    v16h aq0 = load_a_frag(qrow, 0, g);
    v16h aq1 = load_a_frag(qrow, 32, g);

    const v8f z8 = {0.f,0.f,0.f,0.f,0.f,0.f,0.f,0.f};
    v8f acc[4] = {z8, z8, z8, z8};
    float m_run[8], s_run[8];
#pragma unroll
    for (int r = 0; r < 8; ++r) { m_run[r] = -3.0e38f; s_run[r] = 0.f; }

    for (int key0 = 0; key0 < SEQ; key0 += 32) {
        // ---- scores for 32 keys as two 16x16 tiles (hoisted loads)
        v16h bk[2][2];
#pragma unroll
        for (int sub = 0; sub < 2; ++sub) {
            const _Float16* krow = kbase + (size_t)(key0 + sub * 16 + c) * 2304;
            bk[sub][0] = *(const v16h*)(krow + g * 16);
            bk[sub][1] = *(const v16h*)(krow + 32 + g * 16);
        }
        v8f sc[2] = {z8, z8};
#pragma unroll
        for (int sub = 0; sub < 2; ++sub) {
            sc[sub] = wmma_f16(aq0, bk[sub][0], sc[sub]);
            sc[sub] = wmma_f16(aq1, bk[sub][1], sc[sub]);
        }
        int msk0 = mask[b * SEQ + key0 + c];
        int msk1 = mask[b * SEQ + key0 + 16 + c];

        // ---- online softmax update
        float p0[8], p1[8];
#pragma unroll
        for (int r = 0; r < 8; ++r) {
            float s0 = msk0 ? -3.0e38f : sc[0][r] * ATSCALE;
            float s1 = msk1 ? -3.0e38f : sc[1][r] * ATSCALE;
            float mx = fmaxf(s0, s1);
            mx = fmaxf(mx, __shfl_xor(mx, 1));
            mx = fmaxf(mx, __shfl_xor(mx, 2));
            mx = fmaxf(mx, __shfl_xor(mx, 4));
            mx = fmaxf(mx, __shfl_xor(mx, 8));
            float mnew  = fmaxf(m_run[r], mx);
            float alpha = __expf(m_run[r] - mnew);
            float e0 = msk0 ? 0.f : __expf(s0 - mnew);
            float e1 = msk1 ? 0.f : __expf(s1 - mnew);
            float ts = e0 + e1;
            ts += __shfl_xor(ts, 1);
            ts += __shfl_xor(ts, 2);
            ts += __shfl_xor(ts, 4);
            ts += __shfl_xor(ts, 8);
            s_run[r] = s_run[r] * alpha + ts;
            m_run[r] = mnew;
            acc[0][r] *= alpha; acc[1][r] *= alpha;
            acc[2][r] *= alpha; acc[3][r] *= alpha;
            p0[r] = e0; p1[r] = e1;
        }

        // ---- C-layout (row=r+8g, col=c) -> LDS -> A-layout (row=c, K contig)
        __syncthreads();
#pragma unroll
        for (int r = 0; r < 8; ++r) {
            pl[(r + 8 * g) * 32 + c]      = (_Float16)p0[r];
            pl[(r + 8 * g) * 32 + 16 + c] = (_Float16)p1[r];
        }
        __syncthreads();
        v16h ap;
        {
            v8h lo = *(const v8h*)(pl + c * 32 + g * 16);
            v8h hi = *(const v8h*)(pl + c * 32 + g * 16 + 8);
#pragma unroll
            for (int i = 0; i < 8; ++i) { ap[i] = lo[i]; ap[8 + i] = hi[i]; }
        }
        // ---- P(16x32) @ V(32x64): B-fragments contiguous from vT (hoisted)
        v16h bv[4];
#pragma unroll
        for (int nt = 0; nt < 4; ++nt)
            bv[nt] = *(const v16h*)(vtb + (size_t)(nt * 16 + c) * SEQ + key0 + g * 16);
#pragma unroll
        for (int nt = 0; nt < 4; ++nt)
            acc[nt] = wmma_f16(ap, bv[nt], acc[nt]);
    }

#pragma unroll
    for (int nt = 0; nt < 4; ++nt) {
#pragma unroll
        for (int r = 0; r < 8; ++r) {
            int tok = b * SEQ + qt * 16 + r + 8 * g;
            float val = acc[nt][r] / s_run[r];
            y[(size_t)tok * DIMC + h * 64 + nt * 16 + c] = (_Float16)val;
        }
    }
}

// ---------------------------------------------------------------- launch
extern "C" void kernel_launch(void* const* d_in, const int* in_sizes, int n_in,
                              void* d_out, int out_size, void* d_ws, size_t ws_size,
                              hipStream_t stream) {
    const float* x      = (const float*)d_in[0];
    const int*   mask   = (const int*)  d_in[1];
    const float* ln1_g  = (const float*)d_in[2];
    const float* ln1_b  = (const float*)d_in[3];
    const float* qkv_w  = (const float*)d_in[4];
    const float* proj_w = (const float*)d_in[5];
    const float* proj_b = (const float*)d_in[6];
    const float* ln2_g  = (const float*)d_in[7];
    const float* ln2_b  = (const float*)d_in[8];
    const float* fc1_w  = (const float*)d_in[9];
    const float* fc1_b  = (const float*)d_in[10];
    const float* fc2_w  = (const float*)d_in[11];
    const float* fc2_b  = (const float*)d_in[12];
    float* out = (float*)d_out;

    char* ws = (char*)d_ws;
    auto carve = [&](size_t bytes) -> char* {
        char* p = ws;
        ws += (bytes + 255) & ~(size_t)255;
        return p;
    };
    _Float16* h16    = (_Float16*)carve((size_t)TOKENS * DIMC * 2);
    _Float16* qkvw16 = (_Float16*)carve((size_t)3 * DIMC * DIMC * 2);
    _Float16* projw16= (_Float16*)carve((size_t)DIMC * DIMC * 2);
    _Float16* fc1w16 = (_Float16*)carve((size_t)HIDDEN * DIMC * 2);
    _Float16* fc2w16 = (_Float16*)carve((size_t)DIMC * HIDDEN * 2);
    _Float16* qkv16  = (_Float16*)carve((size_t)TOKENS * 3 * DIMC * 2);
    _Float16* vT16   = (_Float16*)carve((size_t)BATCH * HEADS * HDIM * SEQ * 2);
    _Float16* y16    = (_Float16*)carve((size_t)TOKENS * DIMC * 2);
    float*    x1     = (float*)   carve((size_t)TOKENS * DIMC * 4);
    _Float16* h2_16  = (_Float16*)carve((size_t)TOKENS * DIMC * 2);
    _Float16* hid16  = (_Float16*)carve((size_t)TOKENS * HIDDEN * 2);

    auto cgrid = [](int n) { return (n + 255) / 256; };
    cast_f32_to_f16<<<cgrid(3*DIMC*DIMC), 256, 0, stream>>>(qkv_w,  qkvw16, 3*DIMC*DIMC);
    cast_f32_to_f16<<<cgrid(DIMC*DIMC),   256, 0, stream>>>(proj_w, projw16, DIMC*DIMC);
    cast_f32_to_f16<<<cgrid(HIDDEN*DIMC), 256, 0, stream>>>(fc1_w,  fc1w16, HIDDEN*DIMC);
    cast_f32_to_f16<<<cgrid(DIMC*HIDDEN), 256, 0, stream>>>(fc2_w,  fc2w16, DIMC*HIDDEN);

    layernorm_to_f16<<<TOKENS, 256, 0, stream>>>(x, ln1_g, ln1_b, h16);
    gemm_wmma<<<(TOKENS/64)*((3*DIMC)/64), 128, 0, stream>>>(
        h16, qkvw16, nullptr, nullptr, nullptr, qkv16, TOKENS, 3*DIMC, DIMC, 0);
    transpose_v<<<(BATCH*HEADS*HDIM*SEQ)/256, 256, 0, stream>>>(qkv16, vT16);
    attn_wmma<<<BATCH*HEADS*(SEQ/16), 32, 0, stream>>>(qkv16, vT16, mask, y16);
    gemm_wmma<<<(TOKENS/64)*(DIMC/64), 128, 0, stream>>>(
        y16, projw16, proj_b, x, x1, nullptr, TOKENS, DIMC, DIMC, 0);
    layernorm_to_f16<<<TOKENS, 256, 0, stream>>>(x1, ln2_g, ln2_b, h2_16);
    gemm_wmma<<<(TOKENS/64)*(HIDDEN/64), 128, 0, stream>>>(
        h2_16, fc1w16, fc1_b, nullptr, nullptr, hid16, TOKENS, HIDDEN, DIMC, 1);
    gemm_wmma<<<(TOKENS/64)*(DIMC/64), 128, 0, stream>>>(
        hid16, fc2w16, fc2_b, x1, out, nullptr, TOKENS, DIMC, HIDDEN, 0);
}
